// GraphClassifier_17025250361829
// MI455X (gfx1250) — matrix operand order repaired and checked by reading, not compiled, not run on hardware
//
#include <hip/hip_runtime.h>
#include <hip/hip_bf16.h>

typedef __attribute__((ext_vector_type(16))) _Float16 v16h;
typedef __attribute__((ext_vector_type(8)))  _Float16 v8h;
typedef __attribute__((ext_vector_type(4)))  _Float16 v4h;
typedef __attribute__((ext_vector_type(8)))  float    v8f;

#define NEG_SLOPE 0.2f

__device__ __forceinline__ float leakyf(float e) { return e >= 0.f ? e : NEG_SLOPE * e; }

// Monotonic float<->uint mapping so atomicMax(u32) == float max (handles negatives).
__device__ __forceinline__ unsigned fenc(float x) {
    unsigned u = __float_as_uint(x);
    return (u & 0x80000000u) ? ~u : (u | 0x80000000u);
}
__device__ __forceinline__ float fdec(unsigned u) {
    return __uint_as_float((u & 0x80000000u) ? (u & 0x7fffffffu) : ~u);
}

// ---------------- WMMA fragment helpers (gfx1250, wave32) ----------------
// A 16x32 f16: lanes 0-15 row=lane, K = kb..kb+15 with kb=0;
// lanes 16-31 row=lane-16, kb=16.
__device__ __forceinline__ v16h load_a_frag(const _Float16* As, int strideH, int kt, int lane) {
    const int row = lane & 15;
    const int kb  = kt * 32 + ((lane >> 4) ? 16 : 0);
    const _Float16* p = As + row * strideH + kb;
    v16h a;
#pragma unroll
    for (int i = 0; i < 16; ++i) { a[i] = p[i]; }
    return a;
}

// B 32x16 f16 pre-packed in fragment order: frag(nt,kt) = 32 lanes x 16 halves contiguous.
__device__ __forceinline__ v16h load_b_frag(const _Float16* Bp, int KT, int nt, int kt, int lane) {
    const _Float16* p = Bp + (((size_t)(nt * KT + kt)) << 9) + (lane << 4);
    return *(const v16h*)p;  // 32B aligned
}

// Pack weight [K, Nout] (f32 row-major) into per-fragment B layout (f16):
// lanes 0-15: col = nt*16+lane, K = kt*32 + i (i=0..15); lanes 16-31: col = nt*16+lane-16, K = kt*32+16+i.
__global__ void pack_w(const float* __restrict__ Wt, _Float16* __restrict__ out, int K, int Nout) {
    const int KT = K / 32;
    const int total = (Nout / 16) * KT * 512;
    for (int t = blockIdx.x * blockDim.x + threadIdx.x; t < total; t += gridDim.x * blockDim.x) {
        const int frag = t >> 9;
        const int within = t & 511;
        const int lane = within >> 4;
        const int i = within & 15;
        const int nt = frag / KT;
        const int kt = frag - nt * KT;
        const int col = nt * 16 + (lane & 15);
        const int kk  = kt * 32 + ((lane >> 4) << 4) + i;
        out[t] = (_Float16)Wt[(size_t)kk * Nout + col];
    }
}

// ---------------- GEMM 1: xp = x @ W  (N x 128, K=128) ----------------
__global__ void __launch_bounds__(256)
gemm_xp_kernel(const float* __restrict__ x, const _Float16* __restrict__ Wp,
               float* __restrict__ xp, int N) {
    __shared__ _Float16 As[8 * 16 * 136];
    __shared__ float Cs[8][16][20];   // padded: float4-aligned, conflict-free
    const int w = threadIdx.x >> 5, l = threadIdx.x & 31;
    const int row0 = blockIdx.x * 128 + w * 16;
    _Float16* Aw = As + w * 16 * 136;
    // stage A strip (f32 -> f16)
#pragma unroll 4
    for (int r = 0; r < 16; ++r) {
        const int row = row0 + r;
        float4 v = make_float4(0.f, 0.f, 0.f, 0.f);
        if (row < N) v = ((const float4*)(x + (size_t)row * 128))[l];
        _Float16* d = Aw + r * 136 + l * 4;
        d[0] = (_Float16)v.x; d[1] = (_Float16)v.y; d[2] = (_Float16)v.z; d[3] = (_Float16)v.w;
    }
    __syncthreads();
    // preload all A fragments (reused across every nt)
    v16h af[4];
#pragma unroll
    for (int kt = 0; kt < 4; ++kt) af[kt] = load_a_frag(Aw, 136, kt, l);
    // register double-buffered B fragments
    v16h bf[2][4];
#pragma unroll
    for (int kt = 0; kt < 4; ++kt) bf[0][kt] = load_b_frag(Wp, 4, 0, kt, l);

    const int rr = l >> 1, cc = (l & 1) * 8;   // epilogue mapping: 2 lanes per row
    const int orow = row0 + rr;
    const int cr = (l >> 4) ? 8 : 0, ccol = l & 15;
#pragma unroll
    for (int nt = 0; nt < 8; ++nt) {
        if (nt < 7) {
#pragma unroll
            for (int kt = 0; kt < 4; ++kt)
                bf[(nt + 1) & 1][kt] = load_b_frag(Wp, 4, nt + 1, kt, l);
        }
        v8f acc = {};
#pragma unroll
        for (int kt = 0; kt < 4; ++kt)
            acc = __builtin_amdgcn_wmma_f32_16x16x32_f16(false, af[kt], false, bf[nt & 1][kt],
                                                         (short)0, acc, false, false);
        // stage C tile through LDS -> coalesced b128 stores
#pragma unroll
        for (int r = 0; r < 8; ++r) Cs[w][cr + r][ccol] = acc[r];
        float4 o0 = *(const float4*)&Cs[w][rr][cc];
        float4 o1 = *(const float4*)&Cs[w][rr][cc + 4];
        if (orow < N) {
            float* op = xp + (size_t)orow * 128 + nt * 16 + cc;
            ((float4*)op)[0] = o0;
            ((float4*)op)[1] = o1;
        }
    }
}

// ---------------- alpha_s / alpha_d: per-node dot products ----------------
__global__ void __launch_bounds__(256)
alpha_kernel(const float* __restrict__ xp, const float* __restrict__ a_src,
             const float* __restrict__ a_dst, float* __restrict__ as_,
             float* __restrict__ ad_, int N) {
    const int w = threadIdx.x >> 5, l = threadIdx.x & 31;
    const int node = blockIdx.x * 8 + w;
    if (node >= N) return;  // wave-uniform
    float4 v = ((const float4*)(xp + (size_t)node * 128))[l];
    float4 s = ((const float4*)a_src)[l];
    float4 d = ((const float4*)a_dst)[l];
    float ps = v.x * s.x + v.y * s.y + v.z * s.z + v.w * s.w;
    float pd = v.x * d.x + v.y * d.y + v.z * d.z + v.w * d.w;
#pragma unroll
    for (int off = 16; off > 0; off >>= 1) {
        ps += __shfl_down(ps, off, 32);
        pd += __shfl_down(pd, off, 32);
    }
    if (l == 0) { as_[node] = ps; ad_[node] = pd; }
}

// ---------------- init: out = bias (accumulator), emax = -inf, denom = 0 ----------------
__global__ void init_kernel(float* __restrict__ out_xe, const float* __restrict__ bias,
                            unsigned* __restrict__ emax, float* __restrict__ denom, int N) {
    const size_t i = (size_t)blockIdx.x * blockDim.x + threadIdx.x;
    const size_t tot = (size_t)N * 128;
    if (i < tot) out_xe[i] = bias[i & 127];
    if (i < (size_t)N) { emax[i] = fenc(-3.0e38f); denom[i] = 0.f; }
}

// ---------------- edge pass 1: segment max ----------------
__global__ void edge_max_kernel(const int* __restrict__ ei, const float* __restrict__ as_,
                                const float* __restrict__ ad_, unsigned* __restrict__ emax,
                                int E, int N) {
    const int i = blockIdx.x * blockDim.x + threadIdx.x;
    if (i >= E + N) return;
    const int s = (i < E) ? ei[i] : (i - E);
    const int d = (i < E) ? ei[E + i] : (i - E);
    const float e = leakyf(as_[s] + ad_[d]);
    atomicMax(&emax[d], fenc(e));
}

// ---------------- edge pass 2: exp + segment sum ----------------
__global__ void edge_exp_kernel(const int* __restrict__ ei, const float* __restrict__ as_,
                                const float* __restrict__ ad_, const unsigned* __restrict__ emax,
                                float* __restrict__ denom, float* __restrict__ wedge,
                                int E, int N) {
    const int i = blockIdx.x * blockDim.x + threadIdx.x;
    if (i >= E + N) return;
    const int s = (i < E) ? ei[i] : (i - E);
    const int d = (i < E) ? ei[E + i] : (i - E);
    const float e = leakyf(as_[s] + ad_[d]);
    const float wv = __expf(e - fdec(emax[d]));
    wedge[i] = wv;
    atomicAdd(&denom[d], wv);
}

// ---------------- edge pass 3: weighted aggregation (wave per edge) ----------------
__global__ void __launch_bounds__(256)
edge_aggr_kernel(const int* __restrict__ ei, const float* __restrict__ xp,
                 const float* __restrict__ wedge, const float* __restrict__ denom,
                 float* __restrict__ out_xe, int E, int N) {
    const int w = threadIdx.x >> 5, l = threadIdx.x & 31;
    const int i = blockIdx.x * 8 + w;
    if (i >= E + N) return;  // wave-uniform
    const int s = (i < E) ? ei[i] : (i - E);
    const int d = (i < E) ? ei[E + i] : (i - E);
    const float c = wedge[i] / denom[d];
    float4 v = ((const float4*)(xp + (size_t)s * 128))[l];
    float* o = out_xe + (size_t)d * 128 + l * 4;
    atomicAdd(o + 0, c * v.x);
    atomicAdd(o + 1, c * v.y);
    atomicAdd(o + 2, c * v.z);
    atomicAdd(o + 3, c * v.w);
}

// ---------------- GEMM 2: h1 = relu([x, relu(xe)] @ W1 + b1), K=256, out f16 ----------------
__global__ void __launch_bounds__(256)
gemm_h1_kernel(const float* __restrict__ x, const float* __restrict__ xe,
               const _Float16* __restrict__ W1p, const float* __restrict__ b1,
               _Float16* __restrict__ h1, int N) {
    __shared__ _Float16 As[8 * 16 * 264];
    __shared__ float Cs[8][16][20];
    const int w = threadIdx.x >> 5, l = threadIdx.x & 31;
    const int row0 = blockIdx.x * 128 + w * 16;
    _Float16* Aw = As + w * 16 * 264;
    const bool hiHalf = (l >= 16);
    const int cbase = (l & 15) * 8;
#pragma unroll 2
    for (int r = 0; r < 16; ++r) {
        const int row = row0 + r;
        float4 v0 = make_float4(0.f, 0.f, 0.f, 0.f), v1 = v0;
        if (row < N) {
            const float* srcp = hiHalf ? xe : x;
            const float4* p = (const float4*)(srcp + (size_t)row * 128 + cbase);
            v0 = p[0]; v1 = p[1];
            if (hiHalf) {  // relu of x_embedded
                v0.x = fmaxf(v0.x, 0.f); v0.y = fmaxf(v0.y, 0.f);
                v0.z = fmaxf(v0.z, 0.f); v0.w = fmaxf(v0.w, 0.f);
                v1.x = fmaxf(v1.x, 0.f); v1.y = fmaxf(v1.y, 0.f);
                v1.z = fmaxf(v1.z, 0.f); v1.w = fmaxf(v1.w, 0.f);
            }
        }
        _Float16* dp = Aw + r * 264 + (hiHalf ? 128 : 0) + cbase;
        dp[0] = (_Float16)v0.x; dp[1] = (_Float16)v0.y; dp[2] = (_Float16)v0.z; dp[3] = (_Float16)v0.w;
        dp[4] = (_Float16)v1.x; dp[5] = (_Float16)v1.y; dp[6] = (_Float16)v1.z; dp[7] = (_Float16)v1.w;
    }
    __syncthreads();
    // preload all 8 A fragments (reused by every nt); B single-buffered (clause-batched loads)
    v16h af[8];
#pragma unroll
    for (int kt = 0; kt < 8; ++kt) af[kt] = load_a_frag(Aw, 264, kt, l);

    const int rr = l >> 1, cc = (l & 1) * 8;
    const int orow = row0 + rr;
    const int cr = (l >> 4) ? 8 : 0, ccol = l & 15;
#pragma unroll
    for (int nt = 0; nt < 8; ++nt) {
        v16h bf[8];
#pragma unroll
        for (int kt = 0; kt < 8; ++kt) bf[kt] = load_b_frag(W1p, 8, nt, kt, l);
        const float bb = b1[nt * 16 + ccol];
        v8f acc = {};
#pragma unroll
        for (int kt = 0; kt < 8; ++kt)
            acc = __builtin_amdgcn_wmma_f32_16x16x32_f16(false, af[kt], false, bf[kt],
                                                         (short)0, acc, false, false);
#pragma unroll
        for (int r = 0; r < 8; ++r) Cs[w][cr + r][ccol] = fmaxf(acc[r] + bb, 0.f);
        float4 o0 = *(const float4*)&Cs[w][rr][cc];
        float4 o1 = *(const float4*)&Cs[w][rr][cc + 4];
        if (orow < N) {
            v8h hv;
            hv[0] = (_Float16)o0.x; hv[1] = (_Float16)o0.y; hv[2] = (_Float16)o0.z; hv[3] = (_Float16)o0.w;
            hv[4] = (_Float16)o1.x; hv[5] = (_Float16)o1.y; hv[6] = (_Float16)o1.z; hv[7] = (_Float16)o1.w;
            *(v8h*)(h1 + (size_t)orow * 128 + nt * 16 + cc) = hv;
        }
    }
}

// ---------------- GEMM 3: h2 = relu(h1 @ W2 + b2), K=128, Nout=16 ----------------
__global__ void __launch_bounds__(256)
gemm_h2_kernel(const _Float16* __restrict__ h1, const _Float16* __restrict__ W2p,
               const float* __restrict__ b2, float* __restrict__ h2, int N) {
    __shared__ _Float16 As[8 * 16 * 136];
    __shared__ float Cs[8][16][20];
    const int w = threadIdx.x >> 5, l = threadIdx.x & 31;
    const int row0 = blockIdx.x * 128 + w * 16;
    _Float16* Aw = As + w * 16 * 136;
#pragma unroll 4
    for (int r = 0; r < 16; ++r) {
        const int row = row0 + r;
        v4h v = {};
        if (row < N) v = ((const v4h*)(h1 + (size_t)row * 128))[l];
        *(v4h*)(Aw + r * 136 + l * 4) = v;
    }
    __syncthreads();
    v8f acc = {};
#pragma unroll
    for (int kt = 0; kt < 4; ++kt) {
        v16h a = load_a_frag(Aw, 136, kt, l);
        v16h b = load_b_frag(W2p, 4, 0, kt, l);
        acc = __builtin_amdgcn_wmma_f32_16x16x32_f16(false, a, false, b, (short)0, acc, false, false);
    }
    const int cr = (l >> 4) ? 8 : 0, ccol = l & 15;
    const float bb = b2[ccol];
#pragma unroll
    for (int r = 0; r < 8; ++r) Cs[w][cr + r][ccol] = fmaxf(acc[r] + bb, 0.f);
    // coalesced stores: consecutive rows are contiguous in h2 (16 floats/row)
    const int rr = l >> 1, cc = (l & 1) * 8;
    const int orow = row0 + rr;
    float4 o0 = *(const float4*)&Cs[w][rr][cc];
    float4 o1 = *(const float4*)&Cs[w][rr][cc + 4];
    if (orow < N) {
        float* op = h2 + (size_t)orow * 16 + cc;
        ((float4*)op)[0] = o0;
        ((float4*)op)[1] = o1;
    }
}

// ---------------- final: logits = h2 @ W3 + b3; sigmoid ----------------
__global__ void final_kernel(const float* __restrict__ h2, const float* __restrict__ W3,
                             const float* __restrict__ b3, float* __restrict__ probs, int N) {
    const int i = blockIdx.x * blockDim.x + threadIdx.x;
    if (i >= N) return;
    float z0 = b3[0], z1 = b3[1];
    const float* h = h2 + (size_t)i * 16;
#pragma unroll
    for (int k = 0; k < 16; ++k) {
        const float hv = h[k];
        z0 += hv * W3[2 * k];
        z1 += hv * W3[2 * k + 1];
    }
    probs[2 * i]     = 1.f / (1.f + __expf(-z0));
    probs[2 * i + 1] = 1.f / (1.f + __expf(-z1));
}

extern "C" void kernel_launch(void* const* d_in, const int* in_sizes, int n_in,
                              void* d_out, int out_size, void* d_ws, size_t ws_size,
                              hipStream_t stream) {
    const float* x      = (const float*)d_in[0];
    const int*   ei     = (const int*)d_in[1];
    const float* W      = (const float*)d_in[2];
    const float* a_src  = (const float*)d_in[3];
    const float* a_dst  = (const float*)d_in[4];
    const float* bias_g = (const float*)d_in[5];
    const float* W1     = (const float*)d_in[6];
    const float* b1     = (const float*)d_in[7];
    const float* W2     = (const float*)d_in[8];
    const float* b2     = (const float*)d_in[9];
    const float* W3     = (const float*)d_in[10];
    const float* b3     = (const float*)d_in[11];

    const int N = in_sizes[0] / 128;
    const int E = in_sizes[1] / 2;
    const int EN = E + N;

    float* out_xe = (float*)d_out;                   // [N,128] x_embedded
    float* probs  = (float*)d_out + (size_t)N * 128; // [N,2] sigmoid(logits)

    char* ws = (char*)d_ws;
    size_t off = 0;
    auto carve = [&](size_t bytes) -> char* {
        char* p = ws + off;
        off = (off + bytes + 255) & ~(size_t)255;
        return p;
    };
    float*    xp    = (float*)carve((size_t)N * 128 * 4);
    float*    as_   = (float*)carve((size_t)N * 4);
    float*    ad_   = (float*)carve((size_t)N * 4);
    unsigned* emax  = (unsigned*)carve((size_t)N * 4);
    float*    denom = (float*)carve((size_t)N * 4);
    float*    wedge = (float*)carve((size_t)EN * 4);
    _Float16* h1    = (_Float16*)carve((size_t)N * 128 * 2);
    float*    h2    = (float*)carve((size_t)N * 16 * 4);
    _Float16* Wp    = (_Float16*)carve((size_t)128 * 128 * 2);
    _Float16* W1p   = (_Float16*)carve((size_t)256 * 128 * 2);
    _Float16* W2p   = (_Float16*)carve((size_t)128 * 16 * 2);

    // 0) pre-swizzle weights into WMMA B-fragment order (f16)
    pack_w<<<64, 256, 0, stream>>>(W,  Wp,  128, 128);
    pack_w<<<64, 256, 0, stream>>>(W1, W1p, 256, 128);
    pack_w<<<8,  256, 0, stream>>>(W2, W2p, 128, 16);

    const int gblk = (N + 127) / 128;
    // 1) xp = x @ W
    gemm_xp_kernel<<<gblk, 256, 0, stream>>>(x, Wp, xp, N);
    // 2) per-node attention scalars
    alpha_kernel<<<(N + 7) / 8, 256, 0, stream>>>(xp, a_src, a_dst, as_, ad_, N);
    // 3) init accumulators (out = bias_gat)
    {
        const size_t tot = (size_t)N * 128;
        init_kernel<<<(unsigned)((tot + 255) / 256), 256, 0, stream>>>(out_xe, bias_g, emax, denom, N);
    }
    // 4) segment softmax + aggregation over edges (+ self loops)
    edge_max_kernel<<<(EN + 255) / 256, 256, 0, stream>>>(ei, as_, ad_, emax, E, N);
    edge_exp_kernel<<<(EN + 255) / 256, 256, 0, stream>>>(ei, as_, ad_, emax, denom, wedge, E, N);
    edge_aggr_kernel<<<(EN + 7) / 8, 256, 0, stream>>>(ei, xp, wedge, denom, out_xe, E, N);
    // 5) MLP head
    gemm_h1_kernel<<<gblk, 256, 0, stream>>>(x, out_xe, W1p, b1, h1, N);
    gemm_h2_kernel<<<gblk, 256, 0, stream>>>(h1, W2p, b2, h2, N);
    final_kernel<<<(N + 255) / 256, 256, 0, stream>>>(h2, W3, b3, probs, N);
}
